// WindowMHSA_43825846288573
// MI455X (gfx1250) — matrix operand order repaired and checked by reading, not compiled, not run on hardware
//
#include <hip/hip_runtime.h>
#include <math.h>

// ---------------- problem constants ----------------
#define BSZ     4
#define HWDIM   256
#define CCH     256
#define NH      8
#define DH      32
#define WIN     8
#define SHIFT   4
#define NTOK    64          // tokens per window
#define NWIN    4096        // 4 * 32 * 32 windows
#define QKVN    768         // 3*C output cols of qkv gemm
#define QSCALE  0.17677669529663687f   // 32^-0.5

// ---------------- vector types ----------------
typedef _Float16     h16 __attribute__((ext_vector_type(16)));
typedef _Float16     h4  __attribute__((ext_vector_type(4)));
typedef float        f8  __attribute__((ext_vector_type(8)));
typedef unsigned int u4  __attribute__((ext_vector_type(4)));

union FragH { h16 h; u4 u[2]; };

// ---------------- LDS arena layout (bytes) ----------------
// strides in halfs; all byte strides multiples of 16 for b128 loads
#define QS_STR   40      // [64][40] per head  (Q rows, d)
#define VT_STR   72      // [32][72] per head  (Vt: d rows, token)
#define PS_STR   72      // [64][72] per wave  (probs)
#define XS_STR   264     // [64][264]          (x window, f16)
#define OS_STR   264     // [64][264]          (attention output, f16)

#define OFF_Q    0                       // 8*64*40*2  = 40960
#define OFF_K    40960                   // 40960
#define OFF_V    81920                   // 8*32*72*2  = 36864 -> end 118784
#define OFF_X    118784                  // 64*264*2   = 33792 (phase 1 only)
#define OFF_P    118784                  // 8 * 64*72*2 = 73728 (phase 2, overlays X)
#define OFF_O    118784                  // 64*264*2   = 33792 (phase 3, overlays P)
#define OFF_T    192512                  // 225*8*4    = 7200  -> end 199712
#define LDS_BYTES 199712

// ---------------- fragment loaders (ISA 7.12.2 layouts) ----------------
// A 16x32 f16: lanes 0-15 row M=lane hold K {k0..k0+7, k0+16..k0+23},
//              lanes 16-31 row M=lane-16 hold K {k0+8..15, k0+24..31}
__device__ __forceinline__ h16 load_a16(const _Float16* base, int strideH,
                                        int row0, int k0, int m, int half) {
  const _Float16* p = base + (row0 + m) * strideH + k0 + half * 8;
  FragH f;
  f.u[0] = *(const u4*)(p);
  f.u[1] = *(const u4*)(p + 16);
  return f.h;
}
// B 32x16 f16 from N-major storage Bt[n][k]: lane: N = n0+lane%16,
// K = k0 + (lane<16 ? 0 : 8) + {0..7, 16..23} (matches A fragment)
__device__ __forceinline__ h16 load_b16(const _Float16* btBase, int strideH,
                                        int n0, int k0, int m, int half) {
  const _Float16* p = btBase + (n0 + m) * strideH + k0 + half * 8;
  FragH f;
  f.u[0] = *(const u4*)(p);
  f.u[1] = *(const u4*)(p + 16);
  return f.h;
}

#define WMMA_F16(a, b, c) \
  __builtin_amdgcn_wmma_f32_16x16x32_f16(false, (a), false, (b), (short)0, (c), false, false)

__device__ __forceinline__ float half_reduce_max(float v) {
  // butterfly within each 16-lane half (xor<16 never crosses halves)
  #pragma unroll
  for (int off = 1; off < 16; off <<= 1) v = fmaxf(v, __shfl_xor(v, off, 32));
  return v;
}
__device__ __forceinline__ float half_reduce_sum(float v) {
  #pragma unroll
  for (int off = 1; off < 16; off <<= 1) v += __shfl_xor(v, off, 32);
  return v;
}

// ---------------- prep: transpose + f16-convert weights into workspace ----
__global__ void swin_prep_weights(const float* __restrict__ wqkv,
                                  const float* __restrict__ wproj,
                                  _Float16* __restrict__ wt_qkv,   // [768][256]
                                  _Float16* __restrict__ wt_proj)  // [256][256]
{
  int t = blockIdx.x * blockDim.x + threadIdx.x;
  if (t < QKVN * CCH) {
    int n = t / CCH, k = t % CCH;
    wt_qkv[t] = (_Float16)wqkv[k * QKVN + n];
  } else {
    int t2 = t - QKVN * CCH;
    if (t2 < CCH * CCH) {
      int n = t2 / CCH, k = t2 % CCH;
      wt_proj[t2] = (_Float16)wproj[k * CCH + n];
    }
  }
}

// ---------------- fully fused per-window kernel ----------------
// 1 workgroup = 1 window (64 tokens). 256 threads = 8 waves; wave w = head w
// for the attention phase.
__global__ __launch_bounds__(256) void swin_window_mhsa(
    const float*    __restrict__ x,          // [4][256][256][256]
    const float*    __restrict__ b_qkv,      // [768]
    const float*    __restrict__ b_proj,     // [256]
    const float*    __restrict__ bias_table, // [225][8]
    const _Float16* __restrict__ wt_qkv,     // [768][256] (N-major)
    const _Float16* __restrict__ wt_proj,    // [256][256] (N-major)
    float*          __restrict__ out)        // [4][256][256][256]
{
  extern __shared__ __align__(16) char smem[];
  _Float16* Qs  = (_Float16*)(smem + OFF_Q);
  _Float16* Ks  = (_Float16*)(smem + OFF_K);
  _Float16* Vt  = (_Float16*)(smem + OFF_V);
  _Float16* Xs  = (_Float16*)(smem + OFF_X);
  _Float16* Os  = (_Float16*)(smem + OFF_O);
  float*    tab = (float*)   (smem + OFF_T);

  const int tid  = threadIdx.x;
  const int wv   = tid >> 5;          // wave id 0..7
  const int lane = tid & 31;
  const int m    = lane & 15;
  const int half = lane >> 4;

  const int wb = blockIdx.x;          // window index
  const int b  = wb >> 10;
  const int wy = (wb >> 5) & 31;
  const int wx = wb & 31;

  // ---- phase 0: stage shifted x-window (f32->f16) and bias table ----
  for (int t = tid; t < 225 * NH; t += 256) tab[t] = bias_table[t];

  #pragma unroll 4
  for (int it = 0; it < 16; ++it) {
    int chunk = it * 256 + tid;              // 4096 float4 chunks
    int token = chunk >> 6;                  // 64 chunks per 256-ch row
    int c     = (chunk & 63) * 4;
    int iy = token >> 3, ix = token & 7;
    int sy = (wy * WIN + iy + SHIFT) & 255;
    int sx = (wx * WIN + ix + SHIFT) & 255;
    const float4 v = *(const float4*)(x + (((size_t)(b * HWDIM + sy) * HWDIM + sx) * CCH + c));
    h4 hv = { (_Float16)v.x, (_Float16)v.y, (_Float16)v.z, (_Float16)v.w };
    *(h4*)(Xs + token * XS_STR + c) = hv;
  }
  __syncthreads();

  // ---- phase 1: QKV GEMM  [64 x 256] @ [256 x 768] ----
  // each wave owns 6 N-tiles of 16
  for (int j = 0; j < 6; ++j) {
    const int nt = wv * 6 + j;
    const int n0 = nt * 16;
    f8 acc[4];
    #pragma unroll
    for (int mt = 0; mt < 4; ++mt) acc[mt] = (f8){0,0,0,0,0,0,0,0};
    #pragma unroll
    for (int kt = 0; kt < 8; ++kt) {
      const int k0 = kt * 32;
      h16 bf = load_b16(wt_qkv, CCH, n0, k0, m, half);
      #pragma unroll
      for (int mt = 0; mt < 4; ++mt) {
        h16 af = load_a16(Xs, XS_STR, mt * 16, k0, m, half);
        acc[mt] = WMMA_F16(af, bf, acc[mt]);
      }
    }
    // epilogue: +bias, fold q-scale, scatter into Q/K/Vt head layouts
    const int ncol  = n0 + m;
    const float bia = b_qkv[ncol];
    const int which = ncol >> 8;            // 0=Q 1=K 2=V
    const int hc    = ncol & 255;
    const int head  = hc >> 5;
    const int d     = hc & 31;
    const float scl = (which == 0) ? QSCALE : 1.0f;
    #pragma unroll
    for (int mt = 0; mt < 4; ++mt) {
      #pragma unroll
      for (int r = 0; r < 8; ++r) {
        const int token = mt * 16 + r + half * 8;
        const float v = (acc[mt][r] + bia) * scl;
        if (which == 0)      Qs[head * (NTOK * QS_STR) + token * QS_STR + d] = (_Float16)v;
        else if (which == 1) Ks[head * (NTOK * QS_STR) + token * QS_STR + d] = (_Float16)v;
        else                 Vt[head * (DH * VT_STR)   + d * VT_STR + token] = (_Float16)v;
      }
    }
  }
  __syncthreads();

  // ---- phase 2: attention, one head per wave ----
  const int h = wv;
  const _Float16* Qh = Qs + h * (NTOK * QS_STR);
  const _Float16* Kh = Ks + h * (NTOK * QS_STR);
  const _Float16* Vh = Vt + h * (DH * VT_STR);
  _Float16* Pw = (_Float16*)(smem + OFF_P) + wv * (NTOK * PS_STR);

  h16 aQ[4];
  #pragma unroll
  for (int mt = 0; mt < 4; ++mt) aQ[mt] = load_a16(Qh, QS_STR, mt * 16, 0, m, half);

  f8 S[4][4];
  #pragma unroll
  for (int nt = 0; nt < 4; ++nt) {
    h16 bK = load_b16(Kh, QS_STR, nt * 16, 0, m, half);   // K^T fragment
    #pragma unroll
    for (int mt = 0; mt < 4; ++mt) {
      f8 z = (f8){0,0,0,0,0,0,0,0};
      S[mt][nt] = WMMA_F16(aQ[mt], bK, z);
    }
  }

  // + relative position bias
  #pragma unroll
  for (int nt = 0; nt < 4; ++nt) {
    const int col = nt * 16 + m;
    const int cy = col >> 3, cx = col & 7;
    #pragma unroll
    for (int mt = 0; mt < 4; ++mt) {
      #pragma unroll
      for (int r = 0; r < 8; ++r) {
        const int row = mt * 16 + r + half * 8;
        const int ry = row >> 3, rx = row & 7;
        const int idx = (ry - cy + 7) * 15 + (rx - cx + 7);
        S[mt][nt][r] += tab[idx * NH + h];
      }
    }
  }

  // softmax over 64 columns (4 in-lane tiles x 16 lanes/half)
  #pragma unroll
  for (int mt = 0; mt < 4; ++mt) {
    #pragma unroll
    for (int r = 0; r < 8; ++r) {
      float mx = fmaxf(fmaxf(S[mt][0][r], S[mt][1][r]),
                       fmaxf(S[mt][2][r], S[mt][3][r]));
      mx = half_reduce_max(mx);
      float s = 0.f;
      #pragma unroll
      for (int nt = 0; nt < 4; ++nt) {
        float e = __expf(S[mt][nt][r] - mx);
        S[mt][nt][r] = e;
        s += e;
      }
      s = half_reduce_sum(s);
      const float inv = __frcp_rn(s);
      #pragma unroll
      for (int nt = 0; nt < 4; ++nt) {
        const int row = mt * 16 + r + half * 8;
        const int col = nt * 16 + m;
        Pw[row * PS_STR + col] = (_Float16)(S[mt][nt][r] * inv);
      }
    }
  }

  // P @ V  -> O [64 x 32]
  f8 O[4][2];
  #pragma unroll
  for (int mt = 0; mt < 4; ++mt)
    #pragma unroll
    for (int ntd = 0; ntd < 2; ++ntd) O[mt][ntd] = (f8){0,0,0,0,0,0,0,0};

  #pragma unroll
  for (int ntd = 0; ntd < 2; ++ntd) {
    #pragma unroll
    for (int kt = 0; kt < 2; ++kt) {
      h16 bV = load_b16(Vh, VT_STR, ntd * 16, kt * 32, m, half);
      #pragma unroll
      for (int mt = 0; mt < 4; ++mt) {
        h16 aP = load_a16(Pw, PS_STR, mt * 16, kt * 32, m, half);
        O[mt][ntd] = WMMA_F16(aP, bV, O[mt][ntd]);
      }
    }
  }
  __syncthreads();   // everyone done reading P/Q/K/V before Os overlays P

  #pragma unroll
  for (int mt = 0; mt < 4; ++mt)
    #pragma unroll
    for (int ntd = 0; ntd < 2; ++ntd)
      #pragma unroll
      for (int r = 0; r < 8; ++r) {
        const int token = mt * 16 + r + half * 8;
        Os[token * OS_STR + h * DH + ntd * 16 + m] = (_Float16)O[mt][ntd][r];
      }
  __syncthreads();

  // ---- phase 3: output projection [64 x 256] @ [256 x 256], un-shift, store
  for (int j = 0; j < 2; ++j) {
    const int nt = wv * 2 + j;
    const int n0 = nt * 16;
    f8 acc[4];
    #pragma unroll
    for (int mt = 0; mt < 4; ++mt) acc[mt] = (f8){0,0,0,0,0,0,0,0};
    #pragma unroll
    for (int kt = 0; kt < 8; ++kt) {
      const int k0 = kt * 32;
      h16 bf = load_b16(wt_proj, CCH, n0, k0, m, half);
      #pragma unroll
      for (int mt = 0; mt < 4; ++mt) {
        h16 af = load_a16(Os, OS_STR, mt * 16, k0, m, half);
        acc[mt] = WMMA_F16(af, bf, acc[mt]);
      }
    }
    const float bp = b_proj[n0 + m];
    #pragma unroll
    for (int mt = 0; mt < 4; ++mt) {
      #pragma unroll
      for (int r = 0; r < 8; ++r) {
        const int token = mt * 16 + r + half * 8;
        const int iy = token >> 3, ix = token & 7;
        const int sy = (wy * WIN + iy + SHIFT) & 255;   // reverse roll == same map
        const int sx = (wx * WIN + ix + SHIFT) & 255;
        out[((size_t)(b * HWDIM + sy) * HWDIM + sx) * CCH + n0 + m] = acc[mt][r] + bp;
      }
    }
  }
}

// ---------------- host entry ----------------
extern "C" void kernel_launch(void* const* d_in, const int* in_sizes, int n_in,
                              void* d_out, int out_size, void* d_ws, size_t ws_size,
                              hipStream_t stream) {
  const float* x          = (const float*)d_in[0];
  const float* w_qkv      = (const float*)d_in[1];
  const float* b_qkv      = (const float*)d_in[2];
  const float* w_proj     = (const float*)d_in[3];
  const float* b_proj     = (const float*)d_in[4];
  const float* bias_table = (const float*)d_in[5];

  _Float16* wt_qkv  = (_Float16*)d_ws;               // 768*256 f16 = 384 KB
  _Float16* wt_proj = wt_qkv + QKVN * CCH;           // 256*256 f16 = 128 KB

  // one-time (but deterministic, re-run every call) weight transpose+convert
  swin_prep_weights<<<(QKVN * CCH + CCH * CCH + 255) / 256, 256, 0, stream>>>(
      w_qkv, w_proj, wt_qkv, wt_proj);

  // fully fused per-window MHSA: 4096 windows, 8 waves each
  swin_window_mhsa<<<NWIN, 256, LDS_BYTES, stream>>>(
      x, b_qkv, b_proj, bias_table, wt_qkv, wt_proj, (float*)d_out);
}
